// LSTMCell_47347719471523
// MI455X (gfx1250) — matrix operand order repaired and checked by reading, not compiled, not run on hardware
//
#include <hip/hip_runtime.h>
#include <hip/hip_bf16.h>

// ---- CDNA5 (gfx1250) WMMA types ----
typedef __attribute__((ext_vector_type(16))) __bf16 v16bf;
typedef __attribute__((ext_vector_type(8)))  float  v8f;
typedef __attribute__((ext_vector_type(4)))  unsigned int u32x4;

#define BATCH 4096
#define HID   1024
#define KDIM  2048   // input(1024) + hidden(1024)

union FragAB {           // one WMMA A/B operand: 16 bf16 per lane = 8 VGPRs
  v16bf v;
  u32x4 u[2];            // two 16-byte chunks
};

__device__ __forceinline__ unsigned short f2bf(float f) {
  unsigned int u = __float_as_uint(f);
  u += 0x7fffu + ((u >> 16) & 1u);     // round-to-nearest-even
  return (unsigned short)(u >> 16);
}

struct GatePtrs {
  const float* w[4];
  const float* u[4];
  const float* b[4];
};

// ---------------------------------------------------------------------------
// Pack: bf16 X = [incoming | old_h] (4096 x 2048), bf16 V = [W_g | U_g] with
// rows ordered gate-major (4096 x 2048), concatenated bias bcat[4096].
// ---------------------------------------------------------------------------
__global__ __launch_bounds__(256)
void lstm_pack(const float* __restrict__ incoming,
               const float* __restrict__ old_h,
               GatePtrs gp,
               unsigned short* __restrict__ Xbf,
               unsigned short* __restrict__ Vbf,
               float* __restrict__ bcat) {
  const int r = blockIdx.x;     // 0..4095
  const int t = threadIdx.x;    // 0..255
  const int g = r >> 10;
  const int h = r & 1023;
  {
    float4 a = ((const float4*)(incoming + (size_t)r * HID))[t];
    float4 c = ((const float4*)(old_h    + (size_t)r * HID))[t];
    ushort4 pa = make_ushort4(f2bf(a.x), f2bf(a.y), f2bf(a.z), f2bf(a.w));
    ushort4 pc = make_ushort4(f2bf(c.x), f2bf(c.y), f2bf(c.z), f2bf(c.w));
    *(ushort4*)(Xbf + (size_t)r * KDIM + t * 4)        = pa;
    *(ushort4*)(Xbf + (size_t)r * KDIM + 1024 + t * 4) = pc;
  }
  {
    float4 a = ((const float4*)(gp.w[g] + (size_t)h * HID))[t];
    float4 c = ((const float4*)(gp.u[g] + (size_t)h * HID))[t];
    ushort4 pa = make_ushort4(f2bf(a.x), f2bf(a.y), f2bf(a.z), f2bf(a.w));
    ushort4 pc = make_ushort4(f2bf(c.x), f2bf(c.y), f2bf(c.z), f2bf(c.w));
    *(ushort4*)(Vbf + (size_t)r * KDIM + t * 4)        = pa;
    *(ushort4*)(Vbf + (size_t)r * KDIM + 1024 + t * 4) = pc;
  }
  if (t == 0) bcat[r] = gp.b[g][h];
}

// ---------------------------------------------------------------------------
// Fused GEMM (bf16 WMMA) + LSTM pointwise epilogue.
// WG = 128 threads = 4 wave32 waves. WG tile: 128 batch rows x 32 hidden cols
// across all 4 gates. Wave (mi,ni): rows mi*64..+64; ni=0 -> gates i,o;
// ni=1 -> gates f,c. Each wave: 64x64 tile = 4x4 WMMA fragments (16 accums).
// Epilogue: i/o gates -> LDS (column-major, b128 stores); f/c stay in
// registers; ni=1 waves fuse the LSTM pointwise math.
// ---------------------------------------------------------------------------
#define GIO_STRIDE 132   // padded row-count per column (dwords), conflict-free

__global__ __launch_bounds__(128)
void lstm_wmma(const unsigned short* __restrict__ Xbf,
               const unsigned short* __restrict__ Vbf,
               const float* __restrict__ bcat,
               const float* __restrict__ old_c,
               float* __restrict__ new_h,
               float* __restrict__ new_c) {
  __shared__ float gio[64 * GIO_STRIDE];   // 33.8 KB: i at cols 0-31, o at 32-63

  const int tid    = threadIdx.x;
  const int lane   = tid & 31;
  const int wave   = tid >> 5;         // 0..3
  const int mi     = wave >> 1;        // 0..1 : rows mi*64
  const int ni     = wave & 1;         // 0: i,o   1: f,c
  const int l15    = lane & 15;
  const int hiHalf = lane >> 4;        // 0|1

  const int mWG = blockIdx.y * 128;
  const int hWG = blockIdx.x * 32;

  // A operands (16x32 bf16): lanes 0-15 row=l15, K chunks {0..7,16..23};
  // lanes 16-31: {8..15,24..31}
  const unsigned short* aPtr[4];
#pragma unroll
  for (int mf = 0; mf < 4; ++mf) {
    const int row = mWG + mi * 64 + mf * 16 + l15;
    aPtr[mf] = Xbf + (size_t)row * KDIM + (hiHalf ? 8 : 0);
  }
  // B operands (32x16 bf16): lanes 0-15 col=l15, K {0..7,16..23}; lanes 16-31 K {8..15,24..31}
  const unsigned short* bPtr[4];
  float bval[4];
#pragma unroll
  for (int nf = 0; nf < 4; ++nf) {
    const int g   = 2 * ni + (nf >> 1);
    const int col = g * HID + hWG + (nf & 1) * 16 + l15;   // global gate row
    bPtr[nf] = Vbf + (size_t)col * KDIM + (hiHalf ? 8 : 0);
    bval[nf] = bcat[col];            // bias folded into accumulator init
  }

  v8f acc[4][4];
#pragma unroll
  for (int mf = 0; mf < 4; ++mf)
#pragma unroll
    for (int nf = 0; nf < 4; ++nf)
      acc[mf][nf] = (v8f)bval[nf];   // per-column bias splat (C col = lane)

  for (int k = 0; k < KDIM; k += 32) {
    FragAB a[4], b[4];
#pragma unroll
    for (int mf = 0; mf < 4; ++mf) {
      a[mf].u[0] = *(const u32x4*)(aPtr[mf] + k);
      a[mf].u[1] = *(const u32x4*)(aPtr[mf] + k + 16);
    }
#pragma unroll
    for (int nf = 0; nf < 4; ++nf) {
      b[nf].u[0] = *(const u32x4*)(bPtr[nf] + k);
      b[nf].u[1] = *(const u32x4*)(bPtr[nf] + k + 16);
    }
#pragma unroll
    for (int mf = 0; mf < 4; ++mf)
#pragma unroll
      for (int nf = 0; nf < 4; ++nf)
        acc[mf][nf] = __builtin_amdgcn_wmma_f32_16x16x32_bf16(
            false, a[mf].v, false, b[nf].v, (short)0, acc[mf][nf],
            false, false);
  }

  // ---- epilogue ----
  // C layout: VGPR r holds (M = rowBase + r, N = l15); rows contiguous per lane.
  if (ni == 0) {
    // gates i (g=0) and o (g=1) -> LDS column-major: gio[col][row]
#pragma unroll
    for (int mf = 0; mf < 4; ++mf) {
#pragma unroll
      for (int nf = 0; nf < 4; ++nf) {
        const int col     = (nf >> 1) * 32 + (nf & 1) * 16 + l15;  // 0..63
        const int rowBase = mi * 64 + mf * 16 + hiHalf * 8;
        float* p = &gio[col * GIO_STRIDE + rowBase];
        const v8f v = acc[mf][nf];
        *(float4*)(p)     = make_float4(v[0], v[1], v[2], v[3]);
        *(float4*)(p + 4) = make_float4(v[4], v[5], v[6], v[7]);
      }
    }
  }
  __syncthreads();
  if (ni == 1) {
    // own registers hold f (g=2: nf 0,1) and c (g=3: nf 2,3)
#pragma unroll
    for (int mf = 0; mf < 4; ++mf) {
      const int rowBase = mi * 64 + mf * 16 + hiHalf * 8;
#pragma unroll
      for (int hc = 0; hc < 2; ++hc) {           // two h-columns per lane
        const int h  = hc * 16 + l15;            // 0..31
        const int gh = hWG + h;
        const v8f gfv = acc[mf][hc];             // forget gate pre-act
        const v8f gcv = acc[mf][2 + hc];         // cell candidate pre-act
        const float* pi = &gio[h * GIO_STRIDE + rowBase];
        const float* po = &gio[(32 + h) * GIO_STRIDE + rowBase];
#pragma unroll
        for (int r = 0; r < 8; ++r) {
          const float gi = pi[r];
          const float go = po[r];
          const float ig = 1.0f / (1.0f + __expf(-gi));
          const float og = 1.0f / (1.0f + __expf(-go));
          const float fg = 1.0f / (1.0f + __expf(-gfv[r]));
          const float ct = tanhf(gcv[r]);
          const size_t off = (size_t)(mWG + rowBase + r) * HID + gh;
          const float nc = fg * old_c[off] + ig * ct;
          new_c[off] = nc;
          new_h[off] = og * tanhf(nc);
        }
      }
    }
  }
}

// ---------------------------------------------------------------------------
extern "C" void kernel_launch(void* const* d_in, const int* in_sizes, int n_in,
                              void* d_out, int out_size, void* d_ws, size_t ws_size,
                              hipStream_t stream) {
  const float* incoming = (const float*)d_in[0];
  const float* old_h    = (const float*)d_in[1];
  const float* old_c    = (const float*)d_in[2];

  GatePtrs gp;   // gate order in reference: i, o, f, c
  for (int j = 0; j < 4; ++j) {
    gp.w[j] = (const float*)d_in[3 + 3 * j];
    gp.b[j] = (const float*)d_in[4 + 3 * j];
    gp.u[j] = (const float*)d_in[5 + 3 * j];
  }

  unsigned short* Xbf = (unsigned short*)d_ws;                       // 16 MB
  unsigned short* Vbf = Xbf + (size_t)BATCH * KDIM;                  // 16 MB
  float*          bcat = (float*)(Vbf + (size_t)(4 * HID) * KDIM);   // 16 KB

  float* out   = (float*)d_out;
  float* new_h = out;
  float* new_c = out + (size_t)BATCH * HID;

  hipLaunchKernelGGL(lstm_pack, dim3(4096), dim3(256), 0, stream,
                     incoming, old_h, gp, Xbf, Vbf, bcat);
  hipLaunchKernelGGL(lstm_wmma, dim3(HID / 32, BATCH / 128), dim3(128), 0, stream,
                     Xbf, Vbf, bcat, old_c, new_h, new_c);
}
